// PrototypeEnhanced_49203145343524
// MI455X (gfx1250) — matrix operand order repaired and checked by reading, not compiled, not run on hardware
//
#include <hip/hip_runtime.h>
#include <math.h>

// ---------------------------------------------------------------------------
// PrototypeEnhanced block for MI455X (gfx1250, wave32, WMMA).
// Big GEMMs: v_wmma_f32_16x16x32_f16, tiles staged to LDS via the CDNA5
// async global->LDS path (ASYNCcnt) when the builtin is available.
// ---------------------------------------------------------------------------

#define BB   8
#define NT   4096
#define CC   512
#define MM   (BB*NT)     // 32768 token rows
#define HIDN 2048

typedef __attribute__((ext_vector_type(16))) _Float16     v16h;
typedef __attribute__((ext_vector_type(8)))  float        v8f;
typedef __attribute__((ext_vector_type(4)))  unsigned int u32x4;
typedef __attribute__((ext_vector_type(4)))  int          v4i;

union Frag16 { u32x4 q[2]; v16h v; };

#define TILE   64
#define KSTEP  32
#define ROW_DW 20   // 16 data dwords + 4 pad: 80B row stride keeps 16B align, spreads LDS banks

// ---- CDNA5 async global->LDS staging (guarded; falls back to load+ds_store) ----
#if defined(__has_builtin)
#  if __has_builtin(__builtin_amdgcn_global_load_async_to_lds_b128)
#    define ASYNC_LDS 1
#  endif
#endif
#ifndef ASYNC_LDS
#  define ASYNC_LDS 0
#endif

__device__ __forceinline__ void copy16_to_lds(const unsigned int* g, unsigned int* l)
{
#if ASYNC_LDS
  // signature (from toolchain diagnostic): (v4i addrspace(1)* src, lds dst, imm off, cpol)
  __builtin_amdgcn_global_load_async_to_lds_b128(
      (__attribute__((address_space(1))) v4i*)(v4i*)(void*)g,
      (__attribute__((address_space(3))) v4i*)(v4i*)(void*)l, 0, 0);
#else
  *(u32x4*)l = *(const u32x4*)g;
#endif
}

__device__ __forceinline__ void async_wait()
{
#if ASYNC_LDS
#  if __has_builtin(__builtin_amdgcn_s_wait_asynccnt)
  __builtin_amdgcn_s_wait_asynccnt(0);
#  else
  asm volatile("s_wait_asynccnt 0x0" ::: "memory");
#  endif
#endif
}

// ---------------- WMMA GEMM: C[M,Nout] = A[M,K] * W[Nout,K]^T  + epilogue ----
// EPI 0: store f32
// EPI 1: += bias2d[batch(row), col], store f16
// EPI 2: += res[row,col], store f32
// EPI 3: += bias1d[col], exact GELU, store f16
// EPI 4: += bias1d[col] + res[row,col], store f32
template<int EPI>
__launch_bounds__(128)
__global__ void gemm_f16_wmma(const void* __restrict__ Aptr, int a_is_f32,
                              const unsigned int* __restrict__ Wp,   // f16 rows, K/2 dwords each
                              float* __restrict__ outF, _Float16* __restrict__ outH,
                              const float* __restrict__ bias,
                              const float* __restrict__ res,
                              int M, int Nout, int K)
{
  __shared__ __align__(16) unsigned int lA[TILE*ROW_DW];
  __shared__ __align__(16) unsigned int lB[TILE*ROW_DW];
  const int tid  = threadIdx.x;
  const int lane = tid & 31;
  const int wv   = tid >> 5;            // 4 waves, each owns a 64x16 strip
  const int n0   = blockIdx.x * TILE;
  const int m0   = blockIdx.y * TILE;
  const int Kd   = K >> 1;

  v8f acc[4];
#pragma unroll
  for (int t = 0; t < 4; ++t)
#pragma unroll
    for (int r = 0; r < 8; ++r) acc[t][r] = 0.0f;

  const int rowL = tid >> 1;            // 0..63
  const int hlf  = tid & 1;             // which 8-dword half of the 16-dword row
  const unsigned int* A16 = (const unsigned int*)Aptr;
  const float*        A32 = (const float*)Aptr;

  for (int k0 = 0; k0 < K; k0 += KSTEP) {
    // ---- stage A and B tiles into LDS (row-major, K contiguous) ----
    unsigned int* dA = &lA[rowL*ROW_DW + hlf*8];
    if (a_is_f32) {
      const float* s = &A32[(size_t)(m0 + rowL)*K + k0 + hlf*16];
#pragma unroll
      for (int i = 0; i < 8; ++i) {
        union { _Float16 h[2]; unsigned int u; } pk;
        pk.h[0] = (_Float16)s[2*i];
        pk.h[1] = (_Float16)s[2*i+1];
        dA[i] = pk.u;
      }
    } else {
      const unsigned int* s = &A16[(size_t)(m0 + rowL)*Kd + (k0 >> 1) + hlf*8];
      copy16_to_lds(s,     dA);
      copy16_to_lds(s + 4, dA + 4);
    }
    {
      const unsigned int* s = &Wp[(size_t)(n0 + rowL)*Kd + (k0 >> 1) + hlf*8];
      unsigned int* dB = &lB[rowL*ROW_DW + hlf*8];
      copy16_to_lds(s,     dB);
      copy16_to_lds(s + 4, dB + 4);
      if (k0 + KSTEP < K) __builtin_prefetch((const void*)(s + 16), 0, 3);  // global_prefetch_b8
    }
    async_wait();        // this wave's async copies landed in LDS
    __syncthreads();     // all waves' copies (and ds_stores) visible

    // B fragment (32x16 KxN): lane&15 = column n; lanes 16-31 hold K=16..31
    Frag16 bf;
    {
      const unsigned int* p = &lB[(wv*16 + (lane & 15))*ROW_DW + (lane >> 4)*8];
      bf.q[0] = *(const u32x4*)(p);
      bf.q[1] = *(const u32x4*)(p + 4);
    }
#pragma unroll
    for (int t = 0; t < 4; ++t) {
      // A fragment (16x32 MxK): lane&15 = row m; lanes<16 hold K{0..15},
      // lanes>=16 hold K{16..31}  (same contiguous k order as B)
      Frag16 af;
      const unsigned int* p = &lA[(t*16 + (lane & 15))*ROW_DW + (lane >> 4)*8];
      af.q[0] = *(const u32x4*)(p);
      af.q[1] = *(const u32x4*)(p + 4);
      acc[t] = __builtin_amdgcn_wmma_f32_16x16x32_f16(
          false, af.v, false, bf.v, (short)0, acc[t], false, false);
    }
    __syncthreads();
  }

  // ---- epilogue: D layout: vgpr r, lane l -> row = r + 8*(l>>4), col = l&15 ----
  const int col  = n0 + wv*16 + (lane & 15);
  const int rOff = (lane >> 4) * 8;
#pragma unroll
  for (int t = 0; t < 4; ++t) {
#pragma unroll
    for (int r = 0; r < 8; ++r) {
      const int row = m0 + t*16 + rOff + r;
      const size_t idx = (size_t)row * Nout + col;
      float v = acc[t][r];
      if (EPI == 0) {
        outF[idx] = v;
      } else if (EPI == 1) {
        v += bias[(size_t)(row >> 12) * Nout + col];   // 4096 tokens per batch
        outH[idx] = (_Float16)v;
      } else if (EPI == 2) {
        outF[idx] = v + res[idx];
      } else if (EPI == 3) {
        v += bias[col];
        v = 0.5f * v * (1.0f + erff(v * 0.70710678118654752f));  // exact GELU
        outH[idx] = (_Float16)v;
      } else {
        outF[idx] = v + bias[col] + res[idx];
      }
    }
  }
}

// ---------------- LayerNorm over c=512, f32 in -> f16 out -------------------
__launch_bounds__(256)
__global__ void ln_f16_kernel(const float* __restrict__ in, const float* __restrict__ g,
                              const float* __restrict__ beta, _Float16* __restrict__ out)
{
  __shared__ float red[256];
  const int tok = blockIdx.x;
  const int t   = threadIdx.x;
  const float* row = in + (size_t)tok * CC;
  const float a = row[t], b = row[t + 256];
  red[t] = a + b; __syncthreads();
  for (int o = 128; o > 0; o >>= 1) { if (t < o) red[t] += red[t + o]; __syncthreads(); }
  const float mean = red[0] * (1.0f / 512.0f);
  __syncthreads();
  const float d0 = a - mean, d1 = b - mean;
  red[t] = d0*d0 + d1*d1; __syncthreads();
  for (int o = 128; o > 0; o >>= 1) { if (t < o) red[t] += red[t + o]; __syncthreads(); }
  const float rstd = rsqrtf(red[0] * (1.0f / 512.0f) + 1e-5f);
  _Float16* orow = out + (size_t)tok * CC;
  orow[t]       = (_Float16)(d0 * rstd * g[t]       + beta[t]);
  orow[t + 256] = (_Float16)(d1 * rstd * g[t + 256] + beta[t + 256]);
}

// ---------------- weight f32 -> f16 (with row stride / column offset) -------
__global__ void w_to_f16_kernel(const float* __restrict__ src, int stride, int colOff,
                                _Float16* __restrict__ dst, int rows, int cols)
{
  const int i = blockIdx.x * 256 + threadIdx.x;
  if (i >= rows * cols) return;
  const int r = i / cols, cI = i - r * cols;
  dst[i] = (_Float16)src[(size_t)r * stride + colOff + cI];
}

// ---------------- per-batch sum over N (+ EPS_AREA); optional (N - sum) -----
__global__ void rowsum_kernel(const float* __restrict__ w, float* __restrict__ out1,
                              float* __restrict__ out2)
{
  __shared__ float red[256];
  const int b = blockIdx.x, t = threadIdx.x;
  const float* wr = w + (size_t)b * NT;
  float s = 0.f;
  for (int i = t; i < NT; i += 256) s += wr[i];
  red[t] = s; __syncthreads();
  for (int o = 128; o > 0; o >>= 1) { if (t < o) red[t] += red[t + o]; __syncthreads(); }
  if (t == 0) {
    out1[b] = red[0] + 5e-4f;
    if (out2) out2[b] = ((float)NT - red[0]) + 5e-4f;
  }
}

// ---------------- weighted channel pool: out = sum_n w*f (opt: (1-w)*f) -----
__launch_bounds__(128)
__global__ void wpool_kernel(const _Float16* __restrict__ f, const float* __restrict__ w,
                             float* __restrict__ out1, float* __restrict__ out2)
{
  const int b  = blockIdx.y;
  const int ch = blockIdx.x * 128 + threadIdx.x;
  const _Float16* fb = f + (size_t)b * NT * CC + ch;
  const float*    wb = w + (size_t)b * NT;
  float s1 = 0.f, s2 = 0.f;
  for (int n = 0; n < NT; ++n) {
    const float v = (float)fb[(size_t)n * CC];
    const float ww = wb[n];
    s1 += ww * v; s2 += (1.0f - ww) * v;
  }
  out1[b * CC + ch] = s1;
  if (out2) out2[b * CC + ch] = s2;
}

// ---------------- small matvec: out[b,o] = (vin[b,:] . W[o, colOff:+512]) / den[b]
__launch_bounds__(128)
__global__ void matvec_kernel(const float* __restrict__ vin, const float* __restrict__ W,
                              int stride, int colOff, const float* __restrict__ den,
                              float* __restrict__ out, int Nout, int outStride, int outOff)
{
  const int b = blockIdx.y;
  const int o = blockIdx.x * 128 + threadIdx.x;
  if (o >= Nout) return;
  const float* vr = vin + (size_t)b * CC;
  const float* wr = W + (size_t)o * stride + colOff;
  float s = 0.f;
  for (int j = 0; j < CC; ++j) s += vr[j] * wr[j];
  if (den) s /= den[b];
  out[(size_t)b * outStride + outOff + o] = s;
}

// ---------------- per-batch vector L2 norm ----------------------------------
__global__ void vecnorm_kernel(const float* __restrict__ v, float* __restrict__ out)
{
  __shared__ float red[128];
  const int b = blockIdx.x, t = threadIdx.x;
  const float* p = v + (size_t)b * CC;
  float s = 0.f;
  for (int i = t; i < CC; i += 128) s += p[i] * p[i];
  red[t] = s; __syncthreads();
  for (int o = 64; o > 0; o >>= 1) { if (t < o) red[t] += red[t + o]; __syncthreads(); }
  if (t == 0) out[b] = sqrtf(red[0]);
}

// ---------------- per-token cosine sims + 2-way softmax ---------------------
__launch_bounds__(256)
__global__ void sim_softmax_kernel(const float* __restrict__ k,
                                   const float* __restrict__ fg_pro, const float* __restrict__ bg_pro,
                                   const float* __restrict__ nrm_fg, const float* __restrict__ nrm_bg,
                                   float* __restrict__ pseudo_fg, float* __restrict__ out_ps)
{
  __shared__ float rd[3][256];
  const int tok = blockIdx.x;
  const int b = tok >> 12, n = tok & (NT - 1);
  const int t = threadIdx.x;
  const float* kr = k + (size_t)tok * CC;
  const float* fp = fg_pro + (size_t)b * CC;
  const float* bp = bg_pro + (size_t)b * CC;
  const float a0 = kr[t], a1 = kr[t + 256];
  rd[0][t] = a0 * fp[t] + a1 * fp[t + 256];
  rd[1][t] = a0 * bp[t] + a1 * bp[t + 256];
  rd[2][t] = a0 * a0 + a1 * a1;
  __syncthreads();
  for (int o = 128; o > 0; o >>= 1) {
    if (t < o) { rd[0][t] += rd[0][t+o]; rd[1][t] += rd[1][t+o]; rd[2][t] += rd[2][t+o]; }
    __syncthreads();
  }
  if (t == 0) {
    const float nk  = fmaxf(sqrtf(rd[2][0]), 1e-8f);
    const float sfg = rd[0][0] / (nk * fmaxf(nrm_fg[b], 1e-8f)) * 10.0f;
    const float sbg = rd[1][0] / (nk * fmaxf(nrm_bg[b], 1e-8f)) * 10.0f;
    const float mx  = fmaxf(sfg, sbg);
    const float efg = expf(sfg - mx), ebg = expf(sbg - mx);
    const float pfg = efg / (efg + ebg);
    pseudo_fg[tok] = pfg;
    out_ps[(size_t)b * (2*NT) + NT + n] = pfg;        // pseudo[:,1,:]
    out_ps[(size_t)b * (2*NT) + n]      = 1.0f - pfg; // pseudo[:,0,:]
  }
}

// ---------------- pro = s*fg_pro + (1-s)*qf, s = (cos(qf,fg_pro)+1)/2 -------
__launch_bounds__(256)
__global__ void proto_mix_kernel(const float* __restrict__ qf, const float* __restrict__ fg_pro,
                                 const float* __restrict__ nrm_fg, float* __restrict__ pro)
{
  __shared__ float rd[2][256];
  const int b = blockIdx.x, t = threadIdx.x;
  const float* q = qf + (size_t)b * CC;
  const float* p = fg_pro + (size_t)b * CC;
  const float q0 = q[t], q1 = q[t + 256], p0 = p[t], p1 = p[t + 256];
  rd[0][t] = q0 * p0 + q1 * p1;
  rd[1][t] = q0 * q0 + q1 * q1;
  __syncthreads();
  for (int o = 128; o > 0; o >>= 1) {
    if (t < o) { rd[0][t] += rd[0][t+o]; rd[1][t] += rd[1][t+o]; }
    __syncthreads();
  }
  const float nq   = fmaxf(sqrtf(rd[1][0]), 1e-8f);
  const float cosv = rd[0][0] / (nq * fmaxf(nrm_fg[b], 1e-8f));
  const float s    = (cosv + 1.0f) * 0.5f;
  float* pr = pro + (size_t)b * CC;
  pr[t]       = s * p0 + (1.0f - s) * q0;
  pr[t + 256] = s * p1 + (1.0f - s) * q1;
}

// ===========================================================================
extern "C" void kernel_launch(void* const* d_in, const int* in_sizes, int n_in,
                              void* d_out, int out_size, void* d_ws, size_t ws_size,
                              hipStream_t stream)
{
  (void)in_sizes; (void)n_in; (void)out_size; (void)ws_size;
  const float* x     = (const float*)d_in[0];
  const float* y     = (const float*)d_in[1];
  const float* mask  = (const float*)d_in[2];
  const float* prpd  = (const float*)d_in[3];   // pro_pred, (b,1,64,64) == (b,4096)
  const float* g1    = (const float*)d_in[4];
  const float* beta1 = (const float*)d_in[5];
  const float* g2    = (const float*)d_in[6];
  const float* beta2 = (const float*)d_in[7];
  const float* Wq    = (const float*)d_in[8];
  const float* Wk    = (const float*)d_in[9];
  const float* Wv    = (const float*)d_in[10];
  const float* Wx1   = (const float*)d_in[11];
  const float* Wx2   = (const float*)d_in[12];
  const float* Wfx   = (const float*)d_in[13];
  const float* Wy1   = (const float*)d_in[14];
  const float* Wy2   = (const float*)d_in[15];
  const float* Wfy   = (const float*)d_in[16];
  const float* W1x   = (const float*)d_in[17];
  const float* b1x   = (const float*)d_in[18];
  const float* W2x   = (const float*)d_in[19];
  const float* b2x   = (const float*)d_in[20];
  const float* W1y   = (const float*)d_in[21];
  const float* b1y   = (const float*)d_in[22];
  const float* W2y   = (const float*)d_in[23];
  const float* b2y   = (const float*)d_in[24];

  char* ws = (char*)d_ws;
  size_t off = 0;
  auto take = [&](size_t bytes) { size_t o = off; off += (bytes + 255) & ~(size_t)255; return o; };

  const size_t oXN   = take((size_t)MM*CC*2);       // xn f16 -> later ln(x_merge)
  const size_t oYN   = take((size_t)MM*CC*2);       // yn f16 -> later ln(y_merge)
  const size_t oE    = take((size_t)MM*CC*4);       // k f32 -> x_merge -> y_merge
  const size_t oF    = take((size_t)MM*1024*2);     // [x1|x2] f16 -> [y1|y2]
  const size_t oG    = take((size_t)MM*HIDN*2);     // gelu hidden f16 (x then y)
  const size_t oWk   = take((size_t)CC*CC*2);
  const size_t oWx12 = take((size_t)1024*CC*2);
  const size_t oWfx  = take((size_t)CC*1024*2);
  const size_t oW1x  = take((size_t)HIDN*CC*2);
  const size_t oW2x  = take((size_t)CC*HIDN*2);
  const size_t oWy12 = take((size_t)1024*CC*2);
  const size_t oWfy  = take((size_t)CC*1024*2);
  const size_t oW1y  = take((size_t)HIDN*CC*2);
  const size_t oW2y  = take((size_t)CC*HIDN*2);
  const size_t oFGS  = take(BB*CC*4), oBGS = take(BB*CC*4);
  const size_t oDF   = take(BB*4),    oDB  = take(BB*4);
  const size_t oFGP  = take(BB*CC*4), oBGP = take(BB*CC*4);
  const size_t oNF   = take(BB*4),    oNB  = take(BB*4);
  const size_t oPSF  = take((size_t)MM*4);
  const size_t oD1   = take(BB*4),    oD2  = take(BB*4);
  const size_t oVS1  = take(BB*CC*4), oVS2 = take(BB*CC*4);
  const size_t oQ1   = take(BB*CC*4), oQ2  = take(BB*CC*4);
  const size_t oP1   = take(BB*CC*4), oP2  = take(BB*CC*4);
  const size_t oBX   = take((size_t)BB*1024*4);
  const size_t oBY   = take((size_t)BB*1024*4);

  auto F16p = [&](size_t o) { return (_Float16*)(ws + o); };
  auto F32p = [&](size_t o) { return (float*)(ws + o); };
  auto U32p = [&](size_t o) { return (const unsigned int*)(ws + o); };

  // ---- 1) weights -> f16 (extract the K=512 halves of the concat weights) ----
  auto wcvt = [&](const float* src, int stride, int colOff, size_t dst, int rows, int cols) {
    const int n = rows * cols;
    w_to_f16_kernel<<<(n + 255) / 256, 256, 0, stream>>>(src, stride, colOff, F16p(dst), rows, cols);
  };
  wcvt(Wk , CC,   0, oWk,   CC,  CC);
  wcvt(Wx1, 2*CC, 0, oWx12, CC,  CC);
  wcvt(Wx2, 2*CC, 0, oWx12 + (size_t)CC*CC*2, CC, CC);
  wcvt(Wfx, 2*CC, 0, oWfx,  CC,  2*CC);
  wcvt(W1x, CC,   0, oW1x,  HIDN, CC);
  wcvt(W2x, HIDN, 0, oW2x,  CC,  HIDN);
  wcvt(Wy1, 2*CC, 0, oWy12, CC,  CC);
  wcvt(Wy2, 2*CC, 0, oWy12 + (size_t)CC*CC*2, CC, CC);
  wcvt(Wfy, 2*CC, 0, oWfy,  CC,  2*CC);
  wcvt(W1y, CC,   0, oW1y,  HIDN, CC);
  wcvt(W2y, HIDN, 0, oW2y,  CC,  HIDN);

  // ---- 2) layernorm x,y ----
  ln_f16_kernel<<<MM, 256, 0, stream>>>(x, g1, beta1, F16p(oXN));
  ln_f16_kernel<<<MM, 256, 0, stream>>>(y, g1, beta1, F16p(oYN));

  // ---- 3) prototypes (q-GEMM folded: wgap(yn@Wq^T,m) = (sum m*yn)@Wq^T/den) ----
  rowsum_kernel<<<BB, 256, 0, stream>>>(mask, F32p(oDF), F32p(oDB));
  wpool_kernel<<<dim3(CC/128, BB), 128, 0, stream>>>(F16p(oYN), mask, F32p(oFGS), F32p(oBGS));
  matvec_kernel<<<dim3(CC/128, BB), 128, 0, stream>>>(F32p(oFGS), Wq, CC, 0, F32p(oDF), F32p(oFGP), CC, CC, 0);
  matvec_kernel<<<dim3(CC/128, BB), 128, 0, stream>>>(F32p(oBGS), Wq, CC, 0, F32p(oDB), F32p(oBGP), CC, CC, 0);
  vecnorm_kernel<<<BB, 128, 0, stream>>>(F32p(oFGP), F32p(oNF));
  vecnorm_kernel<<<BB, 128, 0, stream>>>(F32p(oBGP), F32p(oNB));

  // ---- 4) k = xn @ Wk^T (WMMA) ----
  gemm_f16_wmma<0><<<dim3(CC/TILE, MM/TILE), 128, 0, stream>>>(
      (const void*)F16p(oXN), 0, U32p(oWk), F32p(oE), nullptr, nullptr, nullptr, MM, CC, CC);

  // ---- 5) cosine sims + softmax -> pseudo (straight into d_out tail) ----
  float* out_ps = (float*)d_out + (size_t)2 * MM * CC;
  sim_softmax_kernel<<<MM, 256, 0, stream>>>(F32p(oE), F32p(oFGP), F32p(oBGP),
                                             F32p(oNF), F32p(oNB), F32p(oPSF), out_ps);

  // ---- 6) v-GEMM folded: query_fg = (sum w*xn)@Wv^T/den; mix prototypes ----
  rowsum_kernel<<<BB, 256, 0, stream>>>(F32p(oPSF), F32p(oD1), nullptr);
  rowsum_kernel<<<BB, 256, 0, stream>>>(prpd,       F32p(oD2), nullptr);
  wpool_kernel<<<dim3(CC/128, BB), 128, 0, stream>>>(F16p(oXN), F32p(oPSF), F32p(oVS1), nullptr);
  wpool_kernel<<<dim3(CC/128, BB), 128, 0, stream>>>(F16p(oXN), prpd,       F32p(oVS2), nullptr);
  matvec_kernel<<<dim3(CC/128, BB), 128, 0, stream>>>(F32p(oVS1), Wv, CC, 0, F32p(oD1), F32p(oQ1), CC, CC, 0);
  matvec_kernel<<<dim3(CC/128, BB), 128, 0, stream>>>(F32p(oVS2), Wv, CC, 0, F32p(oD2), F32p(oQ2), CC, CC, 0);
  proto_mix_kernel<<<BB, 256, 0, stream>>>(F32p(oQ1), F32p(oFGP), F32p(oNF), F32p(oP1));
  proto_mix_kernel<<<BB, 256, 0, stream>>>(F32p(oQ2), F32p(oFGP), F32p(oNF), F32p(oP2));

  // ---- 7) broadcast-prototype halves of concat-GEMMs -> per-batch biases ----
  matvec_kernel<<<dim3(CC/128, BB), 128, 0, stream>>>(F32p(oP1), Wx1, 2*CC, CC, nullptr, F32p(oBX), CC, 1024, 0);
  matvec_kernel<<<dim3(CC/128, BB), 128, 0, stream>>>(F32p(oP2), Wx2, 2*CC, CC, nullptr, F32p(oBX), CC, 1024, CC);
  matvec_kernel<<<dim3(CC/128, BB), 128, 0, stream>>>(F32p(oP1), Wy1, 2*CC, CC, nullptr, F32p(oBY), CC, 1024, 0);
  matvec_kernel<<<dim3(CC/128, BB), 128, 0, stream>>>(F32p(oP2), Wy2, 2*CC, CC, nullptr, F32p(oBY), CC, 1024, CC);

  float* x_out = (float*)d_out;
  float* y_out = (float*)d_out + (size_t)MM * CC;

  // ---- 8) x path: [x1|x2] -> merge(+x) -> LN2 -> GELU-MLP -> x_out ----
  gemm_f16_wmma<1><<<dim3(1024/TILE, MM/TILE), 128, 0, stream>>>(
      (const void*)x, 1, U32p(oWx12), nullptr, F16p(oF), F32p(oBX), nullptr, MM, 1024, CC);
  gemm_f16_wmma<2><<<dim3(CC/TILE, MM/TILE), 128, 0, stream>>>(
      (const void*)F16p(oF), 0, U32p(oWfx), F32p(oE), nullptr, nullptr, x, MM, CC, 1024);
  ln_f16_kernel<<<MM, 256, 0, stream>>>(F32p(oE), g2, beta2, F16p(oXN));
  gemm_f16_wmma<3><<<dim3(HIDN/TILE, MM/TILE), 128, 0, stream>>>(
      (const void*)F16p(oXN), 0, U32p(oW1x), nullptr, F16p(oG), b1x, nullptr, MM, HIDN, CC);
  gemm_f16_wmma<4><<<dim3(CC/TILE, MM/TILE), 128, 0, stream>>>(
      (const void*)F16p(oG), 0, U32p(oW2x), x_out, nullptr, b2x, x, MM, CC, HIDN);

  // ---- 9) y path ----
  gemm_f16_wmma<1><<<dim3(1024/TILE, MM/TILE), 128, 0, stream>>>(
      (const void*)y, 1, U32p(oWy12), nullptr, F16p(oF), F32p(oBY), nullptr, MM, 1024, CC);
  gemm_f16_wmma<2><<<dim3(CC/TILE, MM/TILE), 128, 0, stream>>>(
      (const void*)F16p(oF), 0, U32p(oWfy), F32p(oE), nullptr, nullptr, y, MM, CC, 1024);
  ln_f16_kernel<<<MM, 256, 0, stream>>>(F32p(oE), g2, beta2, F16p(oYN));
  gemm_f16_wmma<3><<<dim3(HIDN/TILE, MM/TILE), 128, 0, stream>>>(
      (const void*)F16p(oYN), 0, U32p(oW1y), nullptr, F16p(oG), b1y, nullptr, MM, HIDN, CC);
  gemm_f16_wmma<4><<<dim3(CC/TILE, MM/TILE), 128, 0, stream>>>(
      (const void*)F16p(oG), 0, U32p(oW2y), y_out, nullptr, b2y, y, MM, CC, HIDN);
}
